// PointerNetWork_11587821764757
// MI455X (gfx1250) — compile-verified, matrix operand order checked by reading, not run on hardware
//
#include <hip/hip_runtime.h>
#include <hip/hip_bf16.h>

typedef __attribute__((ext_vector_type(16))) _Float16 v16h;
typedef __attribute__((ext_vector_type(8)))  float    v8f;

// ---------------- fast math helpers (map to v_exp_f32 / v_rcp_f32 / v_tanh_f32)
__device__ __forceinline__ float fexp(float x){ return __builtin_amdgcn_exp2f(x * 1.44269504088896f); }
__device__ __forceinline__ float frcp(float x){ return __builtin_amdgcn_rcpf(x); }
__device__ __forceinline__ float fsig(float x){ return frcp(1.0f + fexp(-x)); }
#if __has_builtin(__builtin_amdgcn_tanhf)
__device__ __forceinline__ float ftanh(float x){ return __builtin_amdgcn_tanhf(x); }
#else
__device__ __forceinline__ float ftanh(float x){
  x = fminf(fmaxf(x, -15.0f), 15.0f);
  float e = fexp(2.0f * x);
  return (e - 1.0f) * frcp(e + 1.0f);
}
#endif

// D = A(16x32 f16) * B(32x16 f16) + C(16x16 f32)
__device__ __forceinline__ v8f wmma_f16(v16h a, v16h b, v8f c){
  return __builtin_amdgcn_wmma_f32_16x16x32_f16(false, a, false, b, (short)0, c, false, false);
}

// A-fragment LDS layout: element (m, k) of the 16 x 384 x||h tile lives at
// xhA[((k>>5)*32 + m + 16*half)*16 + i], per the ISA 16-bit A layout:
//   K = 32*kc + 16*(i>>3) + 8*half + (i&7), lane = m + 16*half
__device__ __forceinline__ int xh_index(int m, int k){
  int kc = k >> 5, r = k & 31;
  int half = (r >> 3) & 1;
  int i = ((r >> 4) << 3) | (r & 7);
  return (((kc << 5) + m + (half << 4)) << 4) + i;
}

// ---------------- workspace layout (bytes) ----------------------------------
// blend1 : 51200*64 f32                    @ 0          (13,107,200)
// Wg     : 64nt*12kc*32lane*16 f16         @ 13,107,200 (  786,432)
// W2p    : 4*8*32*16 f16                   @ 13,893,632 (   32,768)
// Wfp    : 4*4*32*16 f16                   @ 13,926,400 (   16,384)
// biasg  : 1024 f32 (permuted i/f/g/o)     @ 13,942,784 (    4,096)
// biasf  : 64 f32                          @ 13,946,880 (      256)

// ======== pack W_ih|W_hh (K=384) into B-fragment layout, gate-interleaved ====
// packed column n' = 4*q + p  ->  physical gate row  p*256 + q
__global__ __launch_bounds__(256) void k_pack_gates(
    const float* __restrict__ W_ih, const float* __restrict__ W_hh,
    _Float16* __restrict__ Wg)
{
  int f = blockIdx.x * 256 + threadIdx.x;            // < 393216
  int i = f & 15; int t = f >> 4;
  int lane = t & 31; t >>= 5;
  int kc = t % 12; int nt = t / 12;
  int nprime = nt * 16 + (lane & 15);
  int nphys  = (nprime & 3) * 256 + (nprime >> 2);
  int k = 32 * kc + 16 * (i >> 3) + 8 * (lane >> 4) + (i & 7);
  float v = (k < 128) ? W_ih[nphys * 128 + k] : W_hh[nphys * 256 + (k - 128)];
  Wg[f] = (_Float16)v;
}

__global__ __launch_bounds__(256) void k_pack_w2(
    const float* __restrict__ W2, _Float16* __restrict__ W2p)
{
  int f = blockIdx.x * 256 + threadIdx.x;            // < 16384
  int i = f & 15; int t = f >> 4;
  int lane = t & 31; t >>= 5;
  int kc = t & 7; int nt = t >> 3;
  int n = nt * 16 + (lane & 15);
  int k = 32 * kc + 16 * (i >> 3) + 8 * (lane >> 4) + (i & 7);
  W2p[f] = (_Float16)W2[n * 256 + k];
}

// fused encoder weight: Wf[n][k] = sum_h W1[n][h] * W_enc[h][k], packed f16
__global__ __launch_bounds__(256) void k_pack_wf(
    const float* __restrict__ W1, const float* __restrict__ W_enc,
    _Float16* __restrict__ Wfp)
{
  int f = blockIdx.x * 256 + threadIdx.x;            // < 8192
  int i = f & 15; int t = f >> 4;
  int lane = t & 31; t >>= 5;
  int kc = t & 3; int nt = t >> 2;
  int n = nt * 16 + (lane & 15);
  int k = 32 * kc + 16 * (i >> 3) + 8 * (lane >> 4) + (i & 7);
  float s = 0.0f;
  for (int h = 0; h < 256; ++h) s += W1[n * 256 + h] * W_enc[h * 128 + k];
  Wfp[f] = (_Float16)s;
}

__global__ __launch_bounds__(256) void k_bias(
    const float* __restrict__ b_ih, const float* __restrict__ b_hh,
    const float* __restrict__ W1, const float* __restrict__ b_enc,
    float* __restrict__ biasg, float* __restrict__ biasf)
{
  int t = blockIdx.x * 256 + threadIdx.x;            // < 1024
  int p = t & 3, q = t >> 2;
  biasg[t] = b_ih[p * 256 + q] + b_hh[p * 256 + q];  // permuted to match n'
  if (t < 64){
    float s = 0.0f;
    for (int h = 0; h < 256; ++h) s += W1[t * 256 + h] * b_enc[h];
    biasf[t] = s;
  }
}

// ======== blend1 = targets @ Wf^T + biasf, one 16-row tile per wave ==========
__global__ __launch_bounds__(256) void k_blend1(
    const float* __restrict__ targets, const _Float16* __restrict__ Wfp,
    const float* __restrict__ biasf, float* __restrict__ blend1)
{
  const int wv = threadIdx.x >> 5, lane = threadIdx.x & 31;
  const int tile = blockIdx.x * 8 + wv;              // < 3200
  const int m = lane & 15, half = lane >> 4;
  const float* arow = targets + (tile * 16 + m) * 128;

  v8f acc[4];
  #pragma unroll
  for (int nt = 0; nt < 4; ++nt)
    #pragma unroll
    for (int r = 0; r < 8; ++r) acc[nt][r] = 0.0f;

  #pragma unroll
  for (int kc = 0; kc < 4; ++kc){
    const float* lo = arow + 32 * kc + 8 * half;
    const float* hi = lo + 16;
    v16h a;
    #pragma unroll
    for (int i = 0; i < 8; ++i){ a[i] = (_Float16)lo[i]; a[8 + i] = (_Float16)hi[i]; }
    #pragma unroll
    for (int nt = 0; nt < 4; ++nt){
      v16h b = *(const v16h*)(Wfp + (((nt * 4 + kc) * 32 + lane) << 4));
      acc[nt] = wmma_f16(a, b, acc[nt]);
    }
  }
  const int nl = lane & 15, rh = lane >> 4;
  #pragma unroll
  for (int nt = 0; nt < 4; ++nt){
    float bb = biasf[nt * 16 + nl];
    #pragma unroll
    for (int r = 0; r < 8; ++r)
      blend1[(tile * 16 + r + 8 * rh) * 64 + nt * 16 + nl] = acc[nt][r] + bb;
  }
}

// ======== fused 50-step decoder: one WG (512 thr / 16 waves) per 16 rows =====
__global__ __launch_bounds__(512) void k_decode(
    const float* __restrict__ targets, const float* __restrict__ h0,
    const float* __restrict__ c0, const float* __restrict__ vt,
    const _Float16* __restrict__ Wg, const _Float16* __restrict__ W2p,
    const float* __restrict__ biasgp, const float* __restrict__ blend1,
    float* __restrict__ dout)
{
  __shared__ __align__(32) _Float16 xhA[12 * 32 * 16];  // x||h, A-fragment layout
  __shared__ float cstate[16][256];
  __shared__ float blend2s[16][64];
  __shared__ float attn[16][52];
  __shared__ float biasL[1024];
  __shared__ unsigned long long selmask[16];
  __shared__ int selidx[16];

  const int t = threadIdx.x;
  const int wv = t >> 5, lane = t & 31;
  const int nl = lane & 15, rh = lane >> 4;
  const int row0 = blockIdx.x * 16;

  // ---- init: bias to LDS, h0/c0 load, dec_in = 0 ----
  for (int e = t; e < 1024; e += 512) biasL[e] = biasgp[e];
  for (int e = t; e < 16 * 256; e += 512){
    int m = e >> 8, q = e & 255;
    cstate[m][q] = c0[(row0 + m) * 256 + q];
    xhA[xh_index(m, 128 + q)] = (_Float16)h0[(row0 + m) * 256 + q];
  }
  for (int e = t; e < 16 * 128; e += 512){
    int m = e >> 7, k = e & 127;
    xhA[xh_index(m, k)] = (_Float16)0.0f;
  }
  if (t < 16){ selmask[t] = 0ull; selidx[t] = 0; }
  __syncthreads();

  #pragma unroll 1
  for (int step = 0; step < 50; ++step){
    // ---- LSTM gates: [16 x 384] x [384 x 64 per wave], gate-interleaved ----
    // wave wv owns packed cols n' in [64*wv, 64*wv+64) == q in [16*wv, 16*wv+16)
    v8f acc[4];
    #pragma unroll
    for (int nt = 0; nt < 4; ++nt)
      #pragma unroll
      for (int r = 0; r < 8; ++r) acc[nt][r] = 0.0f;

    #pragma unroll   // fully unrolled: lets the scheduler pipeline B prefetches
    for (int kc = 0; kc < 12; ++kc){
      v16h a = *(const v16h*)&xhA[((kc << 5) + lane) << 4];
      const _Float16* wp = Wg + ((((wv * 4) * 12 + kc) * 32 + lane) << 4);
      #pragma unroll
      for (int nt = 0; nt < 4; ++nt){
        v16h b = *(const v16h*)(wp + nt * (12 * 32 * 16));
        acc[nt] = wmma_f16(a, b, acc[nt]);
      }
    }

    // ---- nonlinearity on own fragments: 4-lane groups hold (i,f,g,o) ----
    #pragma unroll
    for (int nt = 0; nt < 4; ++nt){
      const float bias = biasL[wv * 64 + nt * 16 + nl];
      const int q = (wv * 64 + nt * 16 + (nl & ~3)) >> 2;
      #pragma unroll
      for (int r = 0; r < 8; ++r){
        float v  = acc[nt][r] + bias;
        float iv = __shfl(v, 0, 4);
        float fv = __shfl(v, 1, 4);
        float gv = __shfl(v, 2, 4);
        float ov = __shfl(v, 3, 4);
        int m = r + 8 * rh;
        float cn = fsig(fv) * cstate[m][q] + fsig(iv) * ftanh(gv);
        if ((lane & 3) == 0){
          cstate[m][q] = cn;
          xhA[xh_index(m, 128 + q)] = (_Float16)(fsig(ov) * ftanh(cn));
        }
      }
    }
    __syncthreads();

    // ---- blend2 = h @ W2^T : waves 0..3, one 16-col tile each ----
    if (wv < 4){
      v8f bacc;
      #pragma unroll
      for (int r = 0; r < 8; ++r) bacc[r] = 0.0f;
      #pragma unroll
      for (int kcb = 0; kcb < 8; ++kcb){
        v16h a = *(const v16h*)&xhA[(((4 + kcb) << 5) + lane) << 4];
        v16h b = *(const v16h*)(W2p + (((wv * 8 + kcb) * 32 + lane) << 4));
        bacc = wmma_f16(a, b, bacc);
      }
      #pragma unroll
      for (int r = 0; r < 8; ++r)
        blend2s[r + 8 * rh][wv * 16 + nl] = bacc[r];
    }
    __syncthreads();

    // ---- attention: 32 groups of 16 lanes; group g -> row g>>1, j parity g&1
    {
      const int g = t >> 4, l16 = t & 15;
      const int m = g >> 1, par = g & 1;
      float b2[4], vv[4];
      #pragma unroll
      for (int qd = 0; qd < 4; ++qd){
        b2[qd] = blend2s[m][l16 * 4 + qd];
        vv[qd] = vt[l16 * 4 + qd];
      }
      const float* bp = blend1 + (row0 + m) * 50 * 64 + l16 * 4;
      #pragma unroll 1
      for (int jj = 0; jj < 25; ++jj){
        const int j = par + 2 * jj;
        float4 bv = *(const float4*)(bp + j * 64);
        float s = ftanh(bv.x + b2[0]) * vv[0]
                + ftanh(bv.y + b2[1]) * vv[1]
                + ftanh(bv.z + b2[2]) * vv[2]
                + ftanh(bv.w + b2[3]) * vv[3];
        s += __shfl_xor(s, 1, 16);
        s += __shfl_xor(s, 2, 16);
        s += __shfl_xor(s, 4, 16);
        s += __shfl_xor(s, 8, 16);
        if (l16 == 0){
          if ((selmask[m] >> j) & 1ull) s = -1.0e9f;
          attn[m][j] = s;
        }
      }
    }
    __syncthreads();

    // ---- argmax (first max) + masked softmax + output (threads 0..255) ----
    if (t < 256){
      const int g = t >> 4, l16 = t & 15;
      float vmax = -3.0e38f; int imax = 0;
      float vj[4]; int js[4]; int nj = 0;
      #pragma unroll
      for (int qd = 0; qd < 4; ++qd){
        int j = l16 + 16 * qd;
        if (j < 50){
          float v = attn[g][j];
          vj[nj] = v; js[nj] = j; nj++;
          if (v > vmax){ vmax = v; imax = j; }
        }
      }
      #pragma unroll
      for (int off = 1; off < 16; off <<= 1){
        float ov = __shfl_xor(vmax, off, 16);
        int   oi = __shfl_xor(imax, off, 16);
        if (ov > vmax || (ov == vmax && oi < imax)){ vmax = ov; imax = oi; }
      }
      float s = 0.0f;
      for (int e = 0; e < nj; ++e) s += fexp(vj[e] - vmax);
      s += __shfl_xor(s, 1, 16);
      s += __shfl_xor(s, 2, 16);
      s += __shfl_xor(s, 4, 16);
      s += __shfl_xor(s, 8, 16);
      float inv = frcp(s);
      float* op = dout + ((row0 + g) * 50 + step) * 50;
      for (int e = 0; e < nj; ++e)
        op[js[e]] = fmaxf(fexp(vj[e] - vmax) * inv, 1e-9f);
      if (l16 == 0){ selmask[g] |= (1ull << imax); selidx[g] = imax; }
    }
    __syncthreads();

    // ---- gather next dec_in = targets[row, sel_idx] into A layout ----
    #pragma unroll 1
    for (int e = t; e < 16 * 128; e += 512){
      int m = e >> 7, k = e & 127;
      float v = targets[((row0 + m) * 50 + selidx[m]) * 128 + k];
      xhA[xh_index(m, k)] = (_Float16)v;
    }
    __syncthreads();
  }
}

// ============================================================================
extern "C" void kernel_launch(void* const* d_in, const int* in_sizes, int n_in,
                              void* d_out, int out_size, void* d_ws, size_t ws_size,
                              hipStream_t stream)
{
  (void)in_sizes; (void)n_in; (void)out_size; (void)ws_size;
  const float* targets = (const float*)d_in[0];
  const float* h0    = (const float*)d_in[1];
  const float* c0    = (const float*)d_in[2];
  const float* W_enc = (const float*)d_in[3];
  const float* b_enc = (const float*)d_in[4];
  const float* W_ih  = (const float*)d_in[5];
  const float* W_hh  = (const float*)d_in[6];
  const float* b_ih  = (const float*)d_in[7];
  const float* b_hh  = (const float*)d_in[8];
  const float* W1    = (const float*)d_in[9];
  const float* W2    = (const float*)d_in[10];
  const float* vt    = (const float*)d_in[11];
  float* out = (float*)d_out;

  char* ws = (char*)d_ws;
  float*    blend1 = (float*)(ws + 0);
  _Float16* Wg     = (_Float16*)(ws + 13107200);
  _Float16* W2p    = (_Float16*)(ws + 13893632);
  _Float16* Wfp    = (_Float16*)(ws + 13926400);
  float*    biasg  = (float*)(ws + 13942784);
  float*    biasf  = (float*)(ws + 13946880);

  k_pack_gates<<<1536, 256, 0, stream>>>(W_ih, W_hh, Wg);
  k_pack_w2   <<<  64, 256, 0, stream>>>(W2, W2p);
  k_pack_wf   <<<  32, 256, 0, stream>>>(W1, W_enc, Wfp);
  k_bias      <<<   4, 256, 0, stream>>>(b_ih, b_hh, W1, b_enc, biasg, biasf);
  k_blend1    <<< 400, 256, 0, stream>>>(targets, Wfp, biasf, blend1);
  k_decode    <<<  64, 512, 0, stream>>>(targets, h0, c0, vt, Wg, W2p, biasg, blend1, out);
}
